// Pre_Norm___GQA___Gain___Residual_90709709292034
// MI455X (gfx1250) — matrix operand order repaired and checked by reading, not compiled, not run on hardware
//
#include <hip/hip_runtime.h>
#include <hip/hip_bf16.h>
#include <cstdint>
#include <cstddef>

// Problem constants (from reference)
#define B_    4
#define S_    2048
#define D_    1280
#define NH_   20
#define NKV_  4
#define HD_   64
#define GROUP_ 5
#define MROWS_ (B_ * S_)   // 8192

typedef unsigned short u16;
typedef __attribute__((ext_vector_type(16))) __bf16 v16bf;
typedef __attribute__((ext_vector_type(8)))  float  v8f;

union FragU { v16bf v; uint4 q[2]; };

__device__ __forceinline__ u16 f2bf(float f) {
  uint32_t u = __float_as_uint(f);
  u += 0x7FFFu + ((u >> 16) & 1u);          // round-to-nearest-even
  return (u16)(u >> 16);
}

// ---- WMMA fragment loaders (CDNA5 16x16x32 bf16 layouts, CDNA5 ISA) ----
// A 16x32 bf16: lane l -> row m=l&15; ks = {hf*8+0..7, 16+hf*8+0..7}, hf=l>>4.
__device__ __forceinline__ v16bf load_a_frag(const u16* base, int stride) {
  int lane = threadIdx.x & 31;
  const u16* p = base + (size_t)(lane & 15) * stride + (lane >> 4) * 8;
  FragU a;
  a.q[0] = *(const uint4*)p;
  a.q[1] = *(const uint4*)(p + 16);
  return a.v;
}
// B 32x16 bf16 where B = W^T and W is (N,K) row-major:
// lane l -> col n=l&15, ks = {hf*8+0..7, 16+hf*8+0..7} of W row n.
__device__ __forceinline__ v16bf load_b_frag(const u16* base, int stride) {
  int lane = threadIdx.x & 31;
  const u16* p = base + (size_t)(lane & 15) * stride + (lane >> 4) * 8;
  FragU b;
  b.q[0] = *(const uint4*)p;
  b.q[1] = *(const uint4*)(p + 16);
  return b.v;
}

#define WMMA_BF16(a, b, c) \
  __builtin_amdgcn_wmma_f32_16x16x32_bf16(false, (a), false, (b), (short)0, (c), false, false)

// Async copy 16 bytes global -> LDS (ASYNCcnt-tracked DMA, CDNA5 ISA)
__device__ __forceinline__ void async_b128(uint32_t lds, uint64_t ga) {
  asm volatile("global_load_async_to_lds_b128 %0, %1, off"
               :: "v"(lds), "v"(ga) : "memory");
}

// ---------------- fp32 -> bf16 weight conversion ----------------
__global__ void __launch_bounds__(256) f2bf_kernel(const float* __restrict__ src,
                                                   u16* __restrict__ dst, int n) {
  int i = blockIdx.x * 256 + threadIdx.x;
  if (i < n) dst[i] = f2bf(src[i]);
}

// ---------------- RMSNorm: x (row of 1280) -> h bf16 ----------------
__global__ void __launch_bounds__(256) rmsnorm_kernel(const float* __restrict__ x,
                                                      const float* __restrict__ w,
                                                      u16* __restrict__ h) {
  __shared__ float red[256];
  int row = blockIdx.x;
  const float* xr = x + (size_t)row * D_;
  float ss = 0.0f;
  for (int i = threadIdx.x; i < D_; i += 256) { float v = xr[i]; ss += v * v; }
  red[threadIdx.x] = ss;
  __syncthreads();
  for (int off = 128; off > 0; off >>= 1) {
    if (threadIdx.x < (unsigned)off) red[threadIdx.x] += red[threadIdx.x + off];
    __syncthreads();
  }
  float scale = rsqrtf(red[0] * (1.0f / D_) + 1e-5f);
  u16* hr = h + (size_t)row * D_;
  for (int i = threadIdx.x; i < D_; i += 256) hr[i] = f2bf(xr[i] * scale * w[i]);
}

// ---------------- bf16 WMMA GEMM with async-LDS staging ----------------
// C(MxN) = A(MxK) * W(NxK)^T.  Block = 256 threads = 8 waves (2M x 4N),
// block tile 64(M) x 256(N); each wave computes a 32x64 register tile
// (2x4 wmma tiles -> 8 WMMAs per k-step on shared LDS fragments).
// A/B k-slices (64x32, 256x32 bf16) are DMA'd to LDS with
// global_load_async_to_lds_b128, double-buffered and overlapped with WMMA.
#define GBM 64
#define GBN 256
#define LDS_STRIDE 40   // shorts; 80B padded row -> conflict-free b128 frag reads
#define A_ELEMS (GBM * LDS_STRIDE)
#define B_ELEMS (GBN * LDS_STRIDE)

__global__ void __launch_bounds__(256) gemm_bf16_kernel(
    const u16* __restrict__ A, const u16* __restrict__ W, float* __restrict__ C,
    int M, int N, int K,
    const float* __restrict__ residual, const float* __restrict__ gain) {
  __shared__ __align__(16) u16 As[2][A_ELEMS];
  __shared__ __align__(16) u16 Bs[2][B_ELEMS];

  int tid  = threadIdx.x;
  int lane = tid & 31;
  int wid  = tid >> 5;
  int wm = wid & 1;        // 0..1 : 32-row half of the M tile
  int wn = wid >> 1;       // 0..3 : 64-col quarter of the N tile
  int tileM = blockIdx.y * GBM;
  int tileN = blockIdx.x * GBN;

  // staging assignment: A tile 64 rows x 64B -> thread t copies 16B (row t/4, chunk t%4)
  int arow = tid >> 2, achunk = (tid & 3) * 8;                    // shorts
  const u16* agp = A + (size_t)(tileM + arow) * K + achunk;
  // B tile 256 rows x 64B -> thread t copies row t (4 x 16B)
  const u16* bgp = W + (size_t)(tileN + tid) * K;

  uint32_t alds[2], blds[2];
  alds[0] = (uint32_t)(uintptr_t)&As[0][arow * LDS_STRIDE + achunk];
  alds[1] = (uint32_t)(uintptr_t)&As[1][arow * LDS_STRIDE + achunk];
  blds[0] = (uint32_t)(uintptr_t)&Bs[0][tid * LDS_STRIDE];
  blds[1] = (uint32_t)(uintptr_t)&Bs[1][tid * LDS_STRIDE];

  v8f acc[2][4];
#pragma unroll
  for (int i = 0; i < 2; ++i)
#pragma unroll
    for (int j = 0; j < 4; ++j) {
      v8f z = {0.f,0.f,0.f,0.f,0.f,0.f,0.f,0.f};
      acc[i][j] = z;
    }

  auto stage = [&](int buf, int k) {
    uint64_t ga = (uint64_t)(uintptr_t)agp + (uint64_t)k * 2u;
    async_b128(alds[buf], ga);                         // 16B of A slice
    uint64_t gb = (uint64_t)(uintptr_t)bgp + (uint64_t)k * 2u;
    async_b128(blds[buf] +  0, gb +  0);               // 64B of one W row
    async_b128(blds[buf] + 32, gb + 32);
    async_b128(blds[buf] + 64, gb + 64);
    async_b128(blds[buf] + 96, gb + 96);               // 5 async ops / wave / stage
  };

  auto compute = [&](int buf) {
    v16bf a0 = load_a_frag(&As[buf][(wm * 32 +  0) * LDS_STRIDE], LDS_STRIDE);
    v16bf a1 = load_a_frag(&As[buf][(wm * 32 + 16) * LDS_STRIDE], LDS_STRIDE);
#pragma unroll
    for (int j = 0; j < 4; ++j) {
      v16bf bj = load_b_frag(&Bs[buf][(wn * 64 + j * 16) * LDS_STRIDE], LDS_STRIDE);
      acc[0][j] = WMMA_BF16(a0, bj, acc[0][j]);
      acc[1][j] = WMMA_BF16(a1, bj, acc[1][j]);
    }
  };

  int nk = K / 32;
  stage(0, 0);
  for (int i = 0; i < nk - 1; ++i) {
    stage((i + 1) & 1, (i + 1) * 32);                  // prefetch next slice
    asm volatile("s_wait_asynccnt 0x5" ::: "memory");  // current slice landed
    __syncthreads();
    compute(i & 1);
    __syncthreads();                                   // all reads done before overwrite
  }
  asm volatile("s_wait_asynccnt 0x0" ::: "memory");
  __syncthreads();
  compute((nk - 1) & 1);

  int hf = lane >> 4, nc = lane & 15;
  float g = gain ? gain[0] : 1.0f;
#pragma unroll
  for (int ti = 0; ti < 2; ++ti)
#pragma unroll
    for (int tj = 0; tj < 4; ++tj)
#pragma unroll
      for (int r = 0; r < 8; ++r) {
        int row = tileM + wm * 32 + ti * 16 + r + hf * 8;
        int col = tileN + wn * 64 + tj * 16 + nc;
        size_t idx = (size_t)row * N + col;
        float v = acc[ti][tj][r];
        if (residual) v = g * v + residual[idx];
        C[idx] = v;
      }
}

// ---------------- RoPE on Q: fp32 (B*S,1280) -> bf16 [b][h][s][hd] ----------------
__global__ void __launch_bounds__(256) rope_q_kernel(const float* __restrict__ qtmp,
                                                     u16* __restrict__ qbf) {
  int idx = blockIdx.x * 256 + threadIdx.x;     // over B*S*NH*32 pairs
  int j = idx & 31; int t = idx >> 5;
  int h = t % NH_;  int bs = t / NH_;
  int s = bs % S_;  int bb = bs / S_;
  const float* src = qtmp + (size_t)bs * D_ + h * HD_ + 2 * j;
  float x1 = src[0], x2 = src[1];
  float ang = (float)s * __powf(10000.0f, -(float)(2 * j) / (float)HD_);
  float sn, cs; __sincosf(ang, &sn, &cs);
  u16* dst = qbf + ((size_t)(bb * NH_ + h) * S_ + s) * HD_ + 2 * j;
  dst[0] = f2bf(x1 * cs - x2 * sn);
  dst[1] = f2bf(x1 * sn + x2 * cs);
}

// ---------------- RoPE on K: fp32 (B*S,256) -> bf16 [b][kv][s][hd] ----------------
__global__ void __launch_bounds__(256) rope_k_kernel(const float* __restrict__ ktmp,
                                                     u16* __restrict__ kbf) {
  int idx = blockIdx.x * 256 + threadIdx.x;     // over B*S*NKV*32 pairs
  int j = idx & 31; int t = idx >> 5;
  int kv = t % NKV_; int bs = t / NKV_;
  int s = bs % S_;   int bb = bs / S_;
  const float* src = ktmp + (size_t)bs * (NKV_ * HD_) + kv * HD_ + 2 * j;
  float x1 = src[0], x2 = src[1];
  float ang = (float)s * __powf(10000.0f, -(float)(2 * j) / (float)HD_);
  float sn, cs; __sincosf(ang, &sn, &cs);
  u16* dst = kbf + ((size_t)(bb * NKV_ + kv) * S_ + s) * HD_ + 2 * j;
  dst[0] = f2bf(x1 * cs - x2 * sn);
  dst[1] = f2bf(x1 * sn + x2 * cs);
}

// ---------------- V: fp32 (B*S,256) -> bf16 transposed [b][kv][hd][s] ----------------
__global__ void __launch_bounds__(256) vtrans_kernel(const float* __restrict__ vtmp,
                                                     u16* __restrict__ vbf) {
  int idx = blockIdx.x * 256 + threadIdx.x;     // over B*S*NKV*HD
  int d = idx & 63;  int t = idx >> 6;
  int kv = t % NKV_; int bs = t / NKV_;
  int s = bs % S_;   int bb = bs / S_;
  float v = vtmp[(size_t)bs * (NKV_ * HD_) + kv * HD_ + d];
  vbf[((size_t)(bb * NKV_ + kv) * HD_ + d) * S_ + s] = f2bf(v);
}

// ---------------- Flash-style causal GQA attention ----------------
// 8 waves/block, each wave owns one 16-query tile of one (b,h).
// P is re-laid-out C-frag -> A-frag through padded LDS (stride 40 shorts).
#define P_STRIDE 40
__global__ void __launch_bounds__(256) attn_kernel(const u16* __restrict__ qb,
                                                   const u16* __restrict__ kb,
                                                   const u16* __restrict__ vtb,
                                                   u16* __restrict__ ob) {
  __shared__ __align__(16) u16 lds_p[8][16 * P_STRIDE];
  int lane = threadIdx.x & 31;
  int wid  = threadIdx.x >> 5;
  int gw = blockIdx.x * 8 + wid;
  int qt = gw & (S_ / 16 - 1);                  // 128 query tiles
  int bh = gw >> 7;
  int bb = bh / NH_, h = bh % NH_;
  int kvh = h / GROUP_;

  const u16* Q  = qb  + ((size_t)(bb * NH_ + h) * S_ + qt * 16) * HD_;
  const u16* Kp = kb  + (size_t)(bb * NKV_ + kvh) * S_ * HD_;
  const u16* Vt = vtb + (size_t)(bb * NKV_ + kvh) * HD_ * S_;   // [hd][s]

  v16bf qa0 = load_a_frag(Q, HD_);              // hd 0..31
  v16bf qa1 = load_a_frag(Q + 32, HD_);         // hd 32..63

  v8f o0 = {0.f,0.f,0.f,0.f,0.f,0.f,0.f,0.f};
  v8f o1 = o0, o2 = o0, o3 = o0;
  float m[8], l[8];
#pragma unroll
  for (int r = 0; r < 8; ++r) { m[r] = -3.0e38f; l[r] = 0.0f; }

  int hf = lane >> 4, ncol = lane & 15;
  const float scale = 0.125f;                   // 1/sqrt(64)
  int qend = qt * 16 + 16;

  for (int kb0 = 0; kb0 < qend; kb0 += 32) {
    v8f z = {0.f,0.f,0.f,0.f,0.f,0.f,0.f,0.f};
    v8f s0 = WMMA_BF16(qa1, load_b_frag(Kp + (size_t)kb0 * HD_ + 32, HD_),
             WMMA_BF16(qa0, load_b_frag(Kp + (size_t)kb0 * HD_, HD_), z));
    v8f s1 = WMMA_BF16(qa1, load_b_frag(Kp + (size_t)(kb0 + 16) * HD_ + 32, HD_),
             WMMA_BF16(qa0, load_b_frag(Kp + (size_t)(kb0 + 16) * HD_, HD_), z));

    // scale + causal mask (C layout: vgpr r -> query row qt*16+r+8*hf, key col ncol)
#pragma unroll
    for (int r = 0; r < 8; ++r) {
      int qrow = qt * 16 + r + hf * 8;
      s0[r] = (kb0 + ncol      <= qrow) ? s0[r] * scale : -3.0e38f;
      s1[r] = (kb0 + 16 + ncol <= qrow) ? s1[r] * scale : -3.0e38f;
    }

    // online softmax over this 32-key chunk
#pragma unroll
    for (int r = 0; r < 8; ++r) {
      float cm = fmaxf(s0[r], s1[r]);
      cm = fmaxf(cm, __shfl_xor(cm, 1));
      cm = fmaxf(cm, __shfl_xor(cm, 2));
      cm = fmaxf(cm, __shfl_xor(cm, 4));
      cm = fmaxf(cm, __shfl_xor(cm, 8));        // row lives in 16 lanes of one half
      float mn   = fmaxf(m[r], cm);
      float corr = __expf(m[r] - mn);
      float p0 = __expf(s0[r] - mn);
      float p1 = __expf(s1[r] - mn);
      float rs = p0 + p1;
      rs += __shfl_xor(rs, 1);
      rs += __shfl_xor(rs, 2);
      rs += __shfl_xor(rs, 4);
      rs += __shfl_xor(rs, 8);
      l[r] = l[r] * corr + rs;
      m[r] = mn;
      o0[r] *= corr; o1[r] *= corr; o2[r] *= corr; o3[r] *= corr;
      int prow = r + hf * 8;
      lds_p[wid][prow * P_STRIDE + ncol]      = f2bf(p0);
      lds_p[wid][prow * P_STRIDE + 16 + ncol] = f2bf(p1);
    }

    asm volatile("s_wait_dscnt 0" ::: "memory");   // fence the P bounce
    v16bf pa = load_a_frag(&lds_p[wid][0], P_STRIDE);

    // O += P(16x32) * V(32x64): B frags from V^T rows (contiguous in s)
    o0 = WMMA_BF16(pa, load_b_frag(Vt + 0 * 16 * S_ + kb0, S_), o0);
    o1 = WMMA_BF16(pa, load_b_frag(Vt + 1 * 16 * S_ + kb0, S_), o1);
    o2 = WMMA_BF16(pa, load_b_frag(Vt + 2 * 16 * S_ + kb0, S_), o2);
    o3 = WMMA_BF16(pa, load_b_frag(Vt + 3 * 16 * S_ + kb0, S_), o3);
  }

  // epilogue: divide by row sums, store bf16 into (B*S, 1280) at head column
#pragma unroll
  for (int r = 0; r < 8; ++r) {
    float inv = 1.0f / l[r];
    int srow = qt * 16 + r + hf * 8;
    u16* orow = ob + (size_t)(bb * S_ + srow) * D_ + h * HD_;
    orow[0  + ncol] = f2bf(o0[r] * inv);
    orow[16 + ncol] = f2bf(o1[r] * inv);
    orow[32 + ncol] = f2bf(o2[r] * inv);
    orow[48 + ncol] = f2bf(o3[r] * inv);
  }
}

// ---------------- launch ----------------
extern "C" void kernel_launch(void* const* d_in, const int* in_sizes, int n_in,
                              void* d_out, int out_size, void* d_ws, size_t ws_size,
                              hipStream_t stream) {
  const float* x    = (const float*)d_in[0];
  const float* res  = (const float*)d_in[1];
  const float* nw   = (const float*)d_in[2];
  const float* wq   = (const float*)d_in[3];
  const float* wk   = (const float*)d_in[4];
  const float* wv   = (const float*)d_in[5];
  const float* wo   = (const float*)d_in[6];
  const float* gain = (const float*)d_in[7];
  float* out = (float*)d_out;

  char* ws = (char*)d_ws;
  size_t off = 0;
  auto alloc = [&](size_t bytes) -> void* {
    void* p = ws + off;
    off += (bytes + 255) & ~(size_t)255;
    return p;
  };
  u16*   h_bf  = (u16*)alloc((size_t)MROWS_ * D_ * 2);
  u16*   wq_bf = (u16*)alloc((size_t)D_ * D_ * 2);
  u16*   wk_bf = (u16*)alloc((size_t)(NKV_ * HD_) * D_ * 2);
  u16*   wv_bf = (u16*)alloc((size_t)(NKV_ * HD_) * D_ * 2);
  u16*   wo_bf = (u16*)alloc((size_t)D_ * D_ * 2);
  float* qtmp  = (float*)alloc((size_t)MROWS_ * D_ * 4);
  float* ktmp  = (float*)alloc((size_t)MROWS_ * (NKV_ * HD_) * 4);
  float* vtmp  = (float*)alloc((size_t)MROWS_ * (NKV_ * HD_) * 4);
  u16*   q_bf  = (u16*)alloc((size_t)MROWS_ * D_ * 2);
  u16*   k_bf  = (u16*)alloc((size_t)MROWS_ * (NKV_ * HD_) * 2);
  u16*   vt_bf = (u16*)alloc((size_t)MROWS_ * (NKV_ * HD_) * 2);
  u16*   attn_bf = (u16*)qtmp;   // reuse: qtmp is dead after rope_q

  dim3 blk(256);
  f2bf_kernel<<<(D_ * D_ + 255) / 256, blk, 0, stream>>>(wq, wq_bf, D_ * D_);
  f2bf_kernel<<<(NKV_ * HD_ * D_ + 255) / 256, blk, 0, stream>>>(wk, wk_bf, NKV_ * HD_ * D_);
  f2bf_kernel<<<(NKV_ * HD_ * D_ + 255) / 256, blk, 0, stream>>>(wv, wv_bf, NKV_ * HD_ * D_);
  f2bf_kernel<<<(D_ * D_ + 255) / 256, blk, 0, stream>>>(wo, wo_bf, D_ * D_);

  rmsnorm_kernel<<<MROWS_, blk, 0, stream>>>(x, nw, h_bf);

  gemm_bf16_kernel<<<dim3(D_ / GBN, MROWS_ / GBM), blk, 0, stream>>>(
      h_bf, wq_bf, qtmp, MROWS_, D_, D_, nullptr, nullptr);
  gemm_bf16_kernel<<<dim3((NKV_ * HD_) / GBN, MROWS_ / GBM), blk, 0, stream>>>(
      h_bf, wk_bf, ktmp, MROWS_, NKV_ * HD_, D_, nullptr, nullptr);
  gemm_bf16_kernel<<<dim3((NKV_ * HD_) / GBN, MROWS_ / GBM), blk, 0, stream>>>(
      h_bf, wv_bf, vtmp, MROWS_, NKV_ * HD_, D_, nullptr, nullptr);

  rope_q_kernel<<<(MROWS_ * NH_ * 32) / 256, blk, 0, stream>>>(qtmp, q_bf);
  rope_k_kernel<<<(MROWS_ * NKV_ * 32) / 256, blk, 0, stream>>>(ktmp, k_bf);
  vtrans_kernel<<<(MROWS_ * NKV_ * HD_) / 256, blk, 0, stream>>>(vtmp, vt_bf);

  attn_kernel<<<(B_ * NH_ * (S_ / 16)) / 8, blk, 0, stream>>>(q_bf, k_bf, vt_bf, attn_bf);

  gemm_bf16_kernel<<<dim3(D_ / GBN, MROWS_ / GBM), blk, 0, stream>>>(
      attn_bf, wo_bf, out, MROWS_, D_, D_, res, gain);
}
